// MultiHeadAttention_19756849562281
// MI455X (gfx1250) — matrix operand order repaired and checked by reading, not compiled, not run on hardware
//
#include <hip/hip_runtime.h>

typedef __attribute__((ext_vector_type(16))) __bf16 v16bf;
typedef __attribute__((ext_vector_type(8)))  __bf16 v8bf;
typedef __attribute__((ext_vector_type(8)))  float  v8f;
typedef int v4i_t __attribute__((vector_size(16)));

#define D_MODEL 1024
#define NUM_HEADS 16
#define D_K 64
#define MAX_REL 128
#define BATCH 2
#define SEQ 1024
#define K_DIM 1024
#define N_DIM 1024
#define M_DIM 2048
#define LDP 40   // padded LDS row pitch (elements) for 32-wide K panels

#if defined(__has_builtin)
#if __has_builtin(__builtin_amdgcn_global_load_async_to_lds_b128) && \
    __has_builtin(__builtin_amdgcn_s_wait_asynccnt)
#define HAVE_ASYNC_LDS 1
#endif
#endif

// builtin wants: (v4i addrspace(1)* src, v4i addrspace(3)* lds, Imm off, Imm cpol)
#define AS_G_V4(p) ((__attribute__((address_space(1))) v4i_t*)((void*)(p)))
#define AS_L_V4(p) ((__attribute__((address_space(3))) v4i_t*)((void*)(p)))

// ---------- fragment loaders (contiguous k order: k = 16*(lane/16) + s) ----------
__device__ __forceinline__ v16bf load_a_bf(const __bf16* p) {
  v16bf a;
#pragma unroll
  for (int j = 0; j < 8; ++j) a[j] = p[j];
#pragma unroll
  for (int j = 0; j < 8; ++j) a[j + 8] = p[j + 8];
  return a;
}
__device__ __forceinline__ v16bf load_a_f32(const float* p) {
  v16bf a;
#pragma unroll
  for (int j = 0; j < 8; ++j) a[j] = (__bf16)p[j];
#pragma unroll
  for (int j = 0; j < 8; ++j) a[j + 8] = (__bf16)p[j + 8];
  return a;
}
__device__ __forceinline__ v16bf load_b_bf(const __bf16* p) {
  v16bf b;
#pragma unroll
  for (int j = 0; j < 16; ++j) b[j] = p[j];
  return b;
}
__device__ __forceinline__ v8f wmma_bf16(v16bf a, v16bf b, v8f c) {
  return __builtin_amdgcn_wmma_f32_16x16x32_bf16(false, a, false, b, (short)0, c,
                                                 false, false);
}

// ---------- fp32 -> bf16 streaming conversion (8 elems/thread) ----------
__global__ __launch_bounds__(256) void cvt_kernel(const float* __restrict__ src,
                                                  __bf16* __restrict__ dst, int n) {
  const int i = (blockIdx.x * 256 + threadIdx.x) * 8;
  if (i >= n) return;
  v8bf o;
#pragma unroll
  for (int j = 0; j < 8; ++j) o[j] = (__bf16)src[i + j];
  *(v8bf*)(dst + i) = o;
}

// ---------- unified LDS-staged GEMM:  C = A(bf16) @ B(bf16)^T + bias ----------
// Block: 256 thr (8 waves), block tile 128(M) x 64(N), wave tile 32x32,
// K staged in 32-wide double-buffered LDS panels filled with async copies.
// mode 0: bf16 head-split [b,h,s,d] ; mode 1: bf16 transposed [b,h,d,s] ;
// mode 2: f32 row-major [M,N].
__global__ __launch_bounds__(256) void gemm_kernel(
    const __bf16* __restrict__ A, const __bf16* __restrict__ B,
    const float* __restrict__ bias, __bf16* __restrict__ outb,
    float* __restrict__ outf, int mode) {
  __shared__ __align__(16) __bf16 sA[2][128 * LDP];
  __shared__ __align__(16) __bf16 sB[2][64 * LDP];

  const int t = threadIdx.x;
  const int lane = t & 31;
  const int wave = t >> 5;
  const int wm = wave & 3;   // 4 row sub-tiles of 32
  const int wn = wave >> 2;  // 2 col sub-tiles of 32
  const int m0 = (blockIdx.x >> 4) * 128;
  const int n0 = (blockIdx.x & 15) * 64;

  const int am = lane & 15, akh = (lane >> 4) << 4;
  const int bn = lane & 15, bkb = (lane >> 4) << 4;

  // panel copy: A = 128 rows x 4 chunks(16B), B = 64 rows x 4 chunks
  auto copy_panels = [&](int kt, int buf) {
    const int k0 = kt * 32;
#pragma unroll
    for (int c = t; c < 512; c += 256) {
      const int r = c >> 2, cc = c & 3;
      const __bf16* g = A + (size_t)(m0 + r) * K_DIM + k0 + cc * 8;
      __bf16* l = &sA[buf][r * LDP + cc * 8];
#ifdef HAVE_ASYNC_LDS
      __builtin_amdgcn_global_load_async_to_lds_b128(AS_G_V4(g), AS_L_V4(l), 0, 0);
#else
      *(uint4*)l = *(const uint4*)g;
#endif
    }
    {
      const int c = t;  // exactly 256 chunks
      const int r = c >> 2, cc = c & 3;
      const __bf16* g = B + (size_t)(n0 + r) * K_DIM + k0 + cc * 8;
      __bf16* l = &sB[buf][r * LDP + cc * 8];
#ifdef HAVE_ASYNC_LDS
      __builtin_amdgcn_global_load_async_to_lds_b128(AS_G_V4(g), AS_L_V4(l), 0, 0);
#else
      *(uint4*)l = *(const uint4*)g;
#endif
    }
  };

  v8f acc[2][2] = {};
  copy_panels(0, 0);

  const int KT = K_DIM / 32;  // 32 k-steps
  for (int kt = 0; kt < KT; ++kt) {
    const int cur = kt & 1;
#ifdef HAVE_ASYNC_LDS
    __builtin_amdgcn_s_wait_asynccnt(0);
#endif
    __syncthreads();
    if (kt + 1 < KT) copy_panels(kt + 1, cur ^ 1);

    v16bf af[2], bfr[2];
#pragma unroll
    for (int i = 0; i < 2; ++i)
      af[i] = load_a_bf(&sA[cur][(wm * 32 + i * 16 + am) * LDP + akh]);
#pragma unroll
    for (int j = 0; j < 2; ++j)
      bfr[j] = load_b_bf(&sB[cur][(wn * 32 + j * 16 + bn) * LDP + bkb]);
#pragma unroll
    for (int i = 0; i < 2; ++i)
#pragma unroll
      for (int j = 0; j < 2; ++j)
        acc[i][j] = wmma_bf16(af[i], bfr[j], acc[i][j]);
  }

  const int lane15 = lane & 15;
  const int mbase = (lane >> 4) << 3;
#pragma unroll
  for (int i = 0; i < 2; ++i)
#pragma unroll
    for (int j = 0; j < 2; ++j) {
      const int n = n0 + wn * 32 + j * 16 + lane15;
      const float bv = bias[n];
#pragma unroll
      for (int r = 0; r < 8; ++r) {
        const int m = m0 + wm * 32 + i * 16 + mbase + r;
        const float val = acc[i][j][r] + bv;
        if (mode == 2) {
          outf[(size_t)m * N_DIM + n] = val;
        } else {
          const int bb = m >> 10, s = m & 1023;
          const int h = n >> 6, d = n & 63;
          const size_t idx =
              (mode == 1)
                  ? ((size_t)((bb * NUM_HEADS + h) * D_K + d)) * SEQ + s
                  : ((size_t)((bb * NUM_HEADS + h) * SEQ + s)) * D_K + d;
          outb[idx] = (__bf16)val;
        }
      }
    }
}

// ---------- scores (QK^T/8 + rel bias, mask) + row softmax -> attn f32 ----------
__global__ __launch_bounds__(256) void scores_softmax_kernel(
    const __bf16* __restrict__ Qh, const __bf16* __restrict__ Kh,
    const int* __restrict__ mask, const float* __restrict__ rel_table,
    float* __restrict__ attn) {
  __shared__ float sc[16][SEQ];  // 64 KB
  const int qt = blockIdx.x;     // 0..63
  const int bh = blockIdx.y;     // 0..31
  const int bb = bh >> 4, h = bh & 15;
  const int lane = threadIdx.x & 31;
  const int wave = threadIdx.x >> 5;

  const __bf16* qbase = Qh + ((size_t)bh * SEQ + qt * 16) * D_K;
  const __bf16* kbase = Kh + (size_t)bh * SEQ * D_K;

  const int am = lane & 15, akh = (lane >> 4) << 4;
  const int bn = lane & 15, bkb = (lane >> 4) << 4;
  const int mbase = (lane >> 4) << 3;

  const v16bf a0 = load_a_bf(qbase + am * D_K + akh);
  const v16bf a1 = load_a_bf(qbase + am * D_K + akh + 32);

  for (int i = 0; i < 8; ++i) {
    const int ct = wave * 8 + i;
    const __bf16* kb = kbase + (ct * 16 + bn) * D_K + bkb;
    v8f acc = {};
    acc = wmma_bf16(a0, load_b_bf(kb), acc);
    acc = wmma_bf16(a1, load_b_bf(kb + 32), acc);
    const int kg = ct * 16 + (lane & 15);
#pragma unroll
    for (int r = 0; r < 8; ++r) {
      const int qg = qt * 16 + mbase + r;
      int rel = min(max(kg - qg, -(MAX_REL - 1)), MAX_REL - 1) + (MAX_REL - 1);
      float v = acc[r] * 0.125f + rel_table[rel * NUM_HEADS + h];
      if (mask[((size_t)bb * SEQ + qg) * SEQ + kg] == 0) v = -1e9f;
      sc[mbase + r][kg] = v;
    }
  }
  __syncthreads();

  const int row = threadIdx.x >> 4;
  const int sub = threadIdx.x & 15;
  float mx = -3.4e38f;
  for (int c = sub; c < SEQ; c += 16) mx = fmaxf(mx, sc[row][c]);
#pragma unroll
  for (int off = 8; off >= 1; off >>= 1) mx = fmaxf(mx, __shfl_xor(mx, off, 16));
  float sum = 0.f;
  for (int c = sub; c < SEQ; c += 16) {
    const float e = __expf(sc[row][c] - mx);
    sc[row][c] = e;
    sum += e;
  }
#pragma unroll
  for (int off = 8; off >= 1; off >>= 1) sum += __shfl_xor(sum, off, 16);
  const float inv = 1.f / sum;
  float* arow = attn + ((size_t)bh * SEQ + qt * 16 + row) * SEQ;
  for (int c = sub; c < SEQ; c += 16) arow[c] = sc[row][c] * inv;
}

// ---------- PV GEMM, 2x2 register blocking ----------
__global__ __launch_bounds__(256) void pv_kernel(
    const float* __restrict__ attn, const __bf16* __restrict__ Vt,
    __bf16* __restrict__ ctx) {
  const int lane = threadIdx.x & 31;
  const int wave = threadIdx.x >> 5;
  const int tile = blockIdx.x * 8 + wave;  // 32 bh * 32 mt * 2 nt = 2048
  const int nt = tile & 1;
  const int mt = (tile >> 1) & 31;
  const int bh = tile >> 6;
  const int bb = bh >> 4, h = bh & 15;

  const int am = lane & 15, akh = (lane >> 4) << 4;
  const int bn = lane & 15, bkb = (lane >> 4) << 4;

  const float*  abase = attn + ((size_t)bh * SEQ + mt * 32 + am) * SEQ + akh;
  const __bf16* bbase = Vt + ((size_t)bh * D_K + nt * 32 + bn) * SEQ + bkb;

  v8f acc[2][2] = {};
#pragma unroll 2
  for (int k = 0; k < SEQ; k += 32) {
    v16bf af[2], bfr[2];
#pragma unroll
    for (int i = 0; i < 2; ++i) af[i] = load_a_f32(abase + (size_t)i * 16 * SEQ + k);
#pragma unroll
    for (int j = 0; j < 2; ++j) bfr[j] = load_b_bf(bbase + (size_t)j * 16 * SEQ + k);
#pragma unroll
    for (int i = 0; i < 2; ++i)
#pragma unroll
      for (int j = 0; j < 2; ++j)
        acc[i][j] = wmma_bf16(af[i], bfr[j], acc[i][j]);
  }

  const int mbase = (lane >> 4) << 3;
#pragma unroll
  for (int i = 0; i < 2; ++i)
#pragma unroll
    for (int j = 0; j < 2; ++j) {
      const int d = nt * 32 + j * 16 + (lane & 15);
#pragma unroll
      for (int r = 0; r < 8; ++r) {
        const int q = mt * 32 + i * 16 + mbase + r;
        ctx[((size_t)(bb * SEQ + q)) * D_MODEL + h * D_K + d] = (__bf16)acc[i][j][r];
      }
    }
}

extern "C" void kernel_launch(void* const* d_in, const int* in_sizes, int n_in,
                              void* d_out, int out_size, void* d_ws, size_t ws_size,
                              hipStream_t stream) {
  const float* q    = (const float*)d_in[0];
  const float* k    = (const float*)d_in[1];
  const float* v    = (const float*)d_in[2];
  const int*   mask = (const int*)d_in[3];
  const float* Wq   = (const float*)d_in[4];
  const float* bq   = (const float*)d_in[5];
  const float* Wk   = (const float*)d_in[6];
  const float* bk   = (const float*)d_in[7];
  const float* Wv   = (const float*)d_in[8];
  const float* bv   = (const float*)d_in[9];
  const float* Wo   = (const float*)d_in[10];
  const float* bo   = (const float*)d_in[11];
  const float* rel  = (const float*)d_in[12];

  const size_t XE = (size_t)M_DIM * D_MODEL;  // 2,097,152
  const size_t WE = (size_t)D_MODEL * D_MODEL;
  __bf16* qb  = (__bf16*)d_ws;
  __bf16* kb  = qb + XE;
  __bf16* vb  = kb + XE;
  __bf16* Wqb = vb + XE;
  __bf16* Wkb = Wqb + WE;
  __bf16* Wvb = Wkb + WE;
  __bf16* Wob = Wvb + WE;
  __bf16* Qh  = Wob + WE;
  __bf16* Kh  = Qh + XE;
  __bf16* Vt  = Kh + XE;
  __bf16* ctx = Vt + XE;

  float* out  = (float*)d_out;
  float* attn = out + (size_t)BATCH * SEQ * D_MODEL;

  dim3 blk(256);
  const int cbX = (int)(XE / 8 / 256);  // 1024
  const int cbW = (int)(WE / 8 / 256);  // 512
  cvt_kernel<<<cbX, blk, 0, stream>>>(q, qb, (int)XE);
  cvt_kernel<<<cbX, blk, 0, stream>>>(k, kb, (int)XE);
  cvt_kernel<<<cbX, blk, 0, stream>>>(v, vb, (int)XE);
  cvt_kernel<<<cbW, blk, 0, stream>>>(Wq, Wqb, (int)WE);
  cvt_kernel<<<cbW, blk, 0, stream>>>(Wk, Wkb, (int)WE);
  cvt_kernel<<<cbW, blk, 0, stream>>>(Wv, Wvb, (int)WE);
  cvt_kernel<<<cbW, blk, 0, stream>>>(Wo, Wob, (int)WE);

  // 256 blocks: (M/128)=16 x (N/64)=16
  gemm_kernel<<<256, blk, 0, stream>>>(qb, Wqb, bq, Qh, nullptr, 0);
  gemm_kernel<<<256, blk, 0, stream>>>(kb, Wkb, bk, Kh, nullptr, 0);
  gemm_kernel<<<256, blk, 0, stream>>>(vb, Wvb, bv, Vt, nullptr, 1);
  scores_softmax_kernel<<<dim3(64, 32), blk, 0, stream>>>(Qh, Kh, mask, rel, attn);
  pv_kernel<<<256, blk, 0, stream>>>(attn, Vt, ctx);
  gemm_kernel<<<256, blk, 0, stream>>>(ctx, Wob, bo, nullptr, out, 2);
}
